// Swinv2Attention_3281355014193
// MI455X (gfx1250) — matrix operand order repaired and checked by reading, not compiled, not run on hardware
//
#include <hip/hip_runtime.h>
#include <hip/hip_bf16.h>

// ---------------------------------------------------------------------------
// Swin-v2 window attention for gfx1250 (MI455X).
// Fully fused per-window kernel: QKV proj + cosine attention + output proj,
// bf16 WMMA (v_wmma_f32_16x16x32_bf16) with fp32 softmax/l2norm.
// B=4096 windows, N=64 tokens, C=256, H=8 heads, D=32.
// HBM traffic: read X (256 MB) + write out (256 MB) only; weights/bias L2-hot.
// ---------------------------------------------------------------------------

typedef __attribute__((ext_vector_type(16))) __bf16 v16bf;
typedef __attribute__((ext_vector_type(8)))  float  v8f;

union FragB16 { v16bf v; unsigned int u[8]; };

__device__ __forceinline__ unsigned short f2bf(float f) {
  unsigned int u = __float_as_uint(f);
  u += 0x7FFFu + ((u >> 16) & 1u);          // round-to-nearest-even
  return (unsigned short)(u >> 16);
}
__device__ __forceinline__ unsigned int pk2(float a, float b) {
  return (unsigned int)f2bf(a) | ((unsigned int)f2bf(b) << 16);
}

// reductions across the 16 lanes of a half-wave (xor masks < 16 stay in-half)
__device__ __forceinline__ float hred_add(float v) {
  v += __shfl_xor(v, 1, 32);
  v += __shfl_xor(v, 2, 32);
  v += __shfl_xor(v, 4, 32);
  v += __shfl_xor(v, 8, 32);
  return v;
}
__device__ __forceinline__ float hred_max(float v) {
  v = fmaxf(v, __shfl_xor(v, 1, 32));
  v = fmaxf(v, __shfl_xor(v, 2, 32));
  v = fmaxf(v, __shfl_xor(v, 4, 32));
  v = fmaxf(v, __shfl_xor(v, 8, 32));
  return v;
}

// 16-bit A-matrix 16x32 lane layout (ISA 7.12.2): dword j of lane (ln,lg)
// holds K-pair starting at  (j<4 ? 2j+8g : 16+2(j-4)+8g)  -> dword offset:
__device__ __forceinline__ int a_off(int j, int lg) {
  return (j < 4) ? (j + 4 * lg) : (8 + (j - 4) + 4 * lg);
}

// ---------------------------------------------------------------------------
// Kernel 1a: fp32 -> bf16 weight conversion (wq|wk|wv|wo packed in d_ws)
// ---------------------------------------------------------------------------
__global__ void prep_weights(const float* __restrict__ wq,
                             const float* __restrict__ wk,
                             const float* __restrict__ wv,
                             const float* __restrict__ wo,
                             unsigned short* __restrict__ out) {
  int i = blockIdx.x * 256 + threadIdx.x;           // float4 index, 0..16383
  if (i < 16384) {
    const float4* q4 = (const float4*)wq;
    const float4* k4 = (const float4*)wk;
    const float4* v4 = (const float4*)wv;
    const float4* o4 = (const float4*)wo;
    uint2* dst = (uint2*)out;
    float4 v;
    v = q4[i]; dst[i]         = make_uint2(pk2(v.x, v.y), pk2(v.z, v.w));
    v = k4[i]; dst[16384 + i] = make_uint2(pk2(v.x, v.y), pk2(v.z, v.w));
    v = v4[i]; dst[32768 + i] = make_uint2(pk2(v.x, v.y), pk2(v.z, v.w));
    v = o4[i]; dst[49152 + i] = make_uint2(pk2(v.x, v.y), pk2(v.z, v.w));
  }
}

// ---------------------------------------------------------------------------
// Kernel 1b: CPB MLP -> bias stored TRANSPOSED as [H][key n][query m] fp32,
//            plus per-head logit scales.
// ---------------------------------------------------------------------------
__global__ void prep_bias(const float* __restrict__ coords,      // [225][2]
                          const float* __restrict__ w1,          // [512][2]
                          const float* __restrict__ b1,          // [512]
                          const float* __restrict__ w2,          // [8][512]
                          const int*   __restrict__ rel,         // [64][64]
                          const float* __restrict__ logit_scale, // [8]
                          float* __restrict__ biasT,             // [8][64][64]
                          float* __restrict__ scale) {           // [8]
  __shared__ float table[225 * 8];
  const int t = threadIdx.x;
  if (t < 225) {
    float x0 = coords[2 * t + 0];
    float x1 = coords[2 * t + 1];
    float acc[8];
#pragma unroll
    for (int h = 0; h < 8; ++h) acc[h] = 0.f;
    for (int j = 0; j < 512; ++j) {
      float hv = fmaxf(w1[2 * j] * x0 + w1[2 * j + 1] * x1 + b1[j], 0.f);
#pragma unroll
      for (int h = 0; h < 8; ++h) acc[h] += hv * w2[h * 512 + j];
    }
#pragma unroll
    for (int h = 0; h < 8; ++h) table[t * 8 + h] = acc[h];
  }
  if (t < 8) scale[t] = __expf(fminf(logit_scale[t], 4.6051702f)); // ln(100)
  __syncthreads();
  for (int p = t; p < 64 * 64; p += 256) {       // p = query*64 + key
    int idx = rel[p];
    int tp  = (p & 63) * 64 + (p >> 6);          // transposed: key*64 + query
#pragma unroll
    for (int h = 0; h < 8; ++h)
      biasT[h * 4096 + tp] = 16.f / (1.f + __expf(-table[idx * 8 + h]));
  }
}

// ---------------------------------------------------------------------------
// Per-head projection GEMM: O[64x32] = X[64x256] * W_head^T  (bf16 WMMA)
// ---------------------------------------------------------------------------
__device__ __forceinline__ void proj_gemm(const unsigned int* __restrict__ X32,
                                          const unsigned short* __restrict__ wmat,
                                          int head, int ln, int lg,
                                          v8f acc[4][2]) {
  const unsigned int* W32 = (const unsigned int*)wmat;
  const v8f zero = {0.f, 0.f, 0.f, 0.f, 0.f, 0.f, 0.f, 0.f};
#pragma unroll
  for (int ti = 0; ti < 4; ++ti)
#pragma unroll
    for (int tj = 0; tj < 2; ++tj) acc[ti][tj] = zero;

  for (int kk = 0; kk < 8; ++kk) {                  // K = 256 = 8 * 32
    FragB16 a[4], bb[2];
#pragma unroll
    for (int ti = 0; ti < 4; ++ti) {
      int tok = ti * 16 + ln;
#pragma unroll
      for (int j = 0; j < 8; ++j)
        a[ti].u[j] = X32[tok * 128 + kk * 16 + j + 8 * lg];
    }
#pragma unroll
    for (int tj = 0; tj < 2; ++tj) {
      int col = head * 32 + tj * 16 + ln;           // output channel (W row)
#pragma unroll
      for (int j = 0; j < 8; ++j)
        bb[tj].u[j] = W32[col * 128 + kk * 16 + j + 8 * lg];
    }
#pragma unroll
    for (int ti = 0; ti < 4; ++ti)
#pragma unroll
      for (int tj = 0; tj < 2; ++tj)
        acc[ti][tj] = __builtin_amdgcn_wmma_f32_16x16x32_bf16(
            false, a[ti].v, false, bb[tj].v, (short)0, acc[ti][tj], false, false);
  }
}

// ---------------------------------------------------------------------------
// Kernel 2: fully fused per-window attention.
// One block = one window (8 waves); wave w = head w for QKV+attention,
// then wave w = output-column group w for the final projection.
// ---------------------------------------------------------------------------
__global__ __launch_bounds__(256) void attn_kernel(
    const float* __restrict__ X,               // [4096][64][256]
    const unsigned short* __restrict__ wbf,    // bf16 wq|wk|wv|wo
    const float* __restrict__ bq,
    const float* __restrict__ bv,
    const float* __restrict__ bo,
    const float* __restrict__ biasT,           // [8][key][query]
    const float* __restrict__ scale,           // [8]
    float* __restrict__ Y) {                   // [4096][64][256]
  __shared__ unsigned short ldsX[64 * 256];        // 32 KB: X tile, then ctx
  __shared__ unsigned short ldsStage[8 * 64 * 32]; // 32 KB: per-wave staging

  const int b  = blockIdx.x;
  const int t  = threadIdx.x;
  const int w  = t >> 5;        // wave id == head id
  const int l  = t & 31;
  const int lg = l >> 4;        // half-wave group
  const int ln = l & 15;

  // prefetch this head's bf16 weight tiles (wq|wk|wv|wo, 16 KB each) into L2
  {
    const char* base = (const char*)wbf + (size_t)w * 32 * 512;
#pragma unroll
    for (int mtx = 0; mtx < 4; ++mtx)
#pragma unroll
      for (int c = 0; c < 4; ++c)
        __builtin_prefetch(base + mtx * 131072 + (c * 32 + l) * 128, 0, 3);
  }

  // phase 0: cooperative X tile load fp32 -> bf16 LDS (vectorized)
  {
    const float4* xb4 = (const float4*)(X + (size_t)b * (64 * 256));
    uint2* lds4 = (uint2*)ldsX;
#pragma unroll
    for (int k = 0; k < 16; ++k) {
      int i = t + 256 * k;                      // 4096 float4s total
      float4 v = xb4[i];
      lds4[i] = make_uint2(pk2(v.x, v.y), pk2(v.z, v.w));
    }
  }
  __syncthreads();

  const unsigned int* X32 = (const unsigned int*)ldsX;
  unsigned short* stg = ldsStage + w * (64 * 32);   // wave-private 64x32 tile
  const unsigned int* S32 = (const unsigned int*)stg;

  FragB16 aQ[4], bK[4], bV[2][2];
  v8f acc[4][2];
  const float sc = scale[w];

  // ---- Q = X*Wq_h^T + bq -> l2norm -> *scale -> A-fragments ----
  proj_gemm(X32, wbf, w, ln, lg, acc);
#pragma unroll
  for (int ti = 0; ti < 4; ++ti) {
#pragma unroll
    for (int tj = 0; tj < 2; ++tj) {
      float bqv = bq[w * 32 + tj * 16 + ln];
#pragma unroll
      for (int r = 0; r < 8; ++r) acc[ti][tj][r] += bqv;
    }
#pragma unroll
    for (int r = 0; r < 8; ++r) {
      float ss = acc[ti][0][r] * acc[ti][0][r] + acc[ti][1][r] * acc[ti][1][r];
      ss = hred_add(ss);
      float inv = sc / fmaxf(sqrtf(ss), 1e-12f);
      int m = ti * 16 + r + 8 * lg;
      stg[m * 32 +  0 + ln] = f2bf(acc[ti][0][r] * inv);
      stg[m * 32 + 16 + ln] = f2bf(acc[ti][1][r] * inv);
    }
  }
#pragma unroll
  for (int ti = 0; ti < 4; ++ti) {
    int tok = ti * 16 + ln;
#pragma unroll
    for (int j = 0; j < 8; ++j) aQ[ti].u[j] = S32[tok * 16 + j + 8 * lg];
  }

  // ---- K = X*Wk_h^T -> l2norm -> B-fragments (stage K[token][dim]) ----
  proj_gemm(X32, wbf + 65536, w, ln, lg, acc);
#pragma unroll
  for (int ti = 0; ti < 4; ++ti)
#pragma unroll
    for (int r = 0; r < 8; ++r) {
      float ss = acc[ti][0][r] * acc[ti][0][r] + acc[ti][1][r] * acc[ti][1][r];
      ss = hred_add(ss);
      float inv = 1.f / fmaxf(sqrtf(ss), 1e-12f);
      int m = ti * 16 + r + 8 * lg;
      stg[m * 32 +  0 + ln] = f2bf(acc[ti][0][r] * inv);
      stg[m * 32 + 16 + ln] = f2bf(acc[ti][1][r] * inv);
    }
#pragma unroll
  for (int tjs = 0; tjs < 4; ++tjs) {
    int tok = tjs * 16 + ln;
#pragma unroll
    for (int j = 0; j < 8; ++j) bK[tjs].u[j] = S32[tok * 16 + j + 8 * lg];
  }

  // ---- V = X*Wv_h^T + bv -> stage transposed V^T[dim][token] -> B-frags ----
  proj_gemm(X32, wbf + 131072, w, ln, lg, acc);
#pragma unroll
  for (int ti = 0; ti < 4; ++ti)
#pragma unroll
    for (int tj = 0; tj < 2; ++tj) {
      float bvv = bv[w * 32 + tj * 16 + ln];
      int n = tj * 16 + ln;
#pragma unroll
      for (int r = 0; r < 8; ++r) {
        int m = ti * 16 + r + 8 * lg;
        stg[n * 64 + m] = f2bf(acc[ti][tj][r] + bvv);
      }
    }
#pragma unroll
  for (int ks = 0; ks < 2; ++ks)
#pragma unroll
    for (int tjv = 0; tjv < 2; ++tjv) {
      int dim = tjv * 16 + ln;
#pragma unroll
      for (int j = 0; j < 8; ++j)
        bV[ks][tjv].u[j] = S32[dim * 32 + ks * 16 + j + 8 * lg];
    }

  // ---- scores: WMMA with C initialized to the relative-position bias ----
  v8f S[4][4];
  {
    const float* bh = biasT + w * 4096;          // [key n][query m]
#pragma unroll
    for (int ti = 0; ti < 4; ++ti)
#pragma unroll
      for (int tjs = 0; tjs < 4; ++tjs) {
        int n = tjs * 16 + ln;
        const float4* bp = (const float4*)(bh + n * 64 + ti * 16 + 8 * lg);
        float4 b0 = bp[0], b1 = bp[1];           // 8 contiguous query rows
        v8f c = {b0.x, b0.y, b0.z, b0.w, b1.x, b1.y, b1.z, b1.w};
        S[ti][tjs] = __builtin_amdgcn_wmma_f32_16x16x32_bf16(
            false, aQ[ti].v, false, bK[tjs].v, (short)0, c, false, false);
      }
  }

  // ---- softmax over keys (per query row) ----
#pragma unroll
  for (int ti = 0; ti < 4; ++ti)
#pragma unroll
    for (int r = 0; r < 8; ++r) {
      float mx = fmaxf(fmaxf(S[ti][0][r], S[ti][1][r]),
                       fmaxf(S[ti][2][r], S[ti][3][r]));
      mx = hred_max(mx);
      float sum = 0.f;
#pragma unroll
      for (int tjs = 0; tjs < 4; ++tjs) {
        float e = __expf(S[ti][tjs][r] - mx);
        S[ti][tjs][r] = e;
        sum += e;
      }
      sum = hred_add(sum);
      float isum = 1.f / sum;
#pragma unroll
      for (int tjs = 0; tjs < 4; ++tjs) S[ti][tjs][r] *= isum;
    }

  // ---- ctx = P * V  (stage P half-K at a time through LDS) ----
  v8f C[4][2];
  {
    const v8f zero = {0.f, 0.f, 0.f, 0.f, 0.f, 0.f, 0.f, 0.f};
#pragma unroll
    for (int ti = 0; ti < 4; ++ti)
#pragma unroll
      for (int tjv = 0; tjv < 2; ++tjv) C[ti][tjv] = zero;
  }
#pragma unroll
  for (int ks = 0; ks < 2; ++ks) {
#pragma unroll
    for (int ti = 0; ti < 4; ++ti)
#pragma unroll
      for (int th = 0; th < 2; ++th) {
        int tjs = ks * 2 + th;
#pragma unroll
        for (int r = 0; r < 8; ++r) {
          int m = ti * 16 + r + 8 * lg;
          stg[m * 32 + th * 16 + ln] = f2bf(S[ti][tjs][r]);
        }
      }
#pragma unroll
    for (int ti = 0; ti < 4; ++ti) {
      FragB16 aP;
      int tok = ti * 16 + ln;
#pragma unroll
      for (int j = 0; j < 8; ++j) aP.u[j] = S32[tok * 16 + j + 8 * lg];
#pragma unroll
      for (int tjv = 0; tjv < 2; ++tjv)
        C[ti][tjv] = __builtin_amdgcn_wmma_f32_16x16x32_bf16(
            false, aP.v, false, bV[ks][tjv].v, (short)0, C[ti][tjv], false, false);
    }
  }

  // ---- reuse the X tile as the bf16 ctx tile (X is dead after the V GEMM) --
  __syncthreads();                               // all waves done reading X
#pragma unroll
  for (int ti = 0; ti < 4; ++ti)
#pragma unroll
    for (int tjv = 0; tjv < 2; ++tjv) {
      int n = w * 32 + tjv * 16 + ln;
#pragma unroll
      for (int r = 0; r < 8; ++r) {
        int m = ti * 16 + r + 8 * lg;
        ldsX[m * 256 + n] = f2bf(C[ti][tjv][r]);
      }
    }
  __syncthreads();                               // full ctx tile visible

  // ---- output projection: Y = ctx @ Wo^T + bo (wave w -> cols w*32..) ----
  {
    const unsigned int* W32 = (const unsigned int*)(wbf + 196608);
    v8f o[4][2];
    const v8f zero = {0.f, 0.f, 0.f, 0.f, 0.f, 0.f, 0.f, 0.f};
#pragma unroll
    for (int ti = 0; ti < 4; ++ti)
#pragma unroll
      for (int tj = 0; tj < 2; ++tj) o[ti][tj] = zero;

    for (int kk = 0; kk < 8; ++kk) {
      FragB16 a[4], bb[2];
#pragma unroll
      for (int ti = 0; ti < 4; ++ti) {
        int tok = ti * 16 + ln;
#pragma unroll
        for (int j = 0; j < 8; ++j)
          a[ti].u[j] = X32[tok * 128 + kk * 16 + j + 8 * lg];
      }
#pragma unroll
      for (int tj = 0; tj < 2; ++tj) {
        int col = w * 32 + tj * 16 + ln;
#pragma unroll
        for (int j = 0; j < 8; ++j)
          bb[tj].u[j] = W32[col * 128 + kk * 16 + j + 8 * lg];
      }
#pragma unroll
      for (int ti = 0; ti < 4; ++ti)
#pragma unroll
        for (int tj = 0; tj < 2; ++tj)
          o[ti][tj] = __builtin_amdgcn_wmma_f32_16x16x32_bf16(
              false, a[ti].v, false, bb[tj].v, (short)0, o[ti][tj], false, false);
    }

    float* yb = Y + (size_t)b * (64 * 256);
#pragma unroll
    for (int ti = 0; ti < 4; ++ti)
#pragma unroll
      for (int tj = 0; tj < 2; ++tj) {
        int n = w * 32 + tj * 16 + ln;
        float bon = bo[n];
#pragma unroll
        for (int r = 0; r < 8; ++r) {
          int m = ti * 16 + r + 8 * lg;
          yb[m * 256 + n] = o[ti][tj][r] + bon;
        }
      }
  }
}

// ---------------------------------------------------------------------------
extern "C" void kernel_launch(void* const* d_in, const int* in_sizes, int n_in,
                              void* d_out, int out_size, void* d_ws, size_t ws_size,
                              hipStream_t stream) {
  (void)in_sizes; (void)n_in; (void)out_size; (void)ws_size;

  const float* X           = (const float*)d_in[0];
  const float* wq          = (const float*)d_in[1];
  const float* bq          = (const float*)d_in[2];
  const float* wk          = (const float*)d_in[3];
  const float* wv          = (const float*)d_in[4];
  const float* bv          = (const float*)d_in[5];
  const float* wo          = (const float*)d_in[6];
  const float* bo          = (const float*)d_in[7];
  const float* logit_scale = (const float*)d_in[8];
  const float* cpb_w1      = (const float*)d_in[9];
  const float* cpb_b1      = (const float*)d_in[10];
  const float* cpb_w2      = (const float*)d_in[11];
  const float* coords      = (const float*)d_in[12];
  const int*   rel         = (const int*)d_in[13];

  // workspace layout (bytes):
  //   [0, 524288)       : bf16 weights wq|wk|wv|wo   (4 * 65536 ushort)
  //   [524288, 655360)  : fp32 biasT [8][key][query]
  //   [655360, 655392)  : fp32 per-head scales [8]
  unsigned short* wbf  = (unsigned short*)d_ws;
  float* biasbuf       = (float*)((char*)d_ws + 524288);
  float* scalebuf      = (float*)((char*)d_ws + 655360);
  float* out           = (float*)d_out;

  prep_weights<<<64, 256, 0, stream>>>(wq, wk, wv, wo, wbf);
  prep_bias<<<1, 256, 0, stream>>>(coords, cpb_w1, cpb_b1, cpb_w2, rel,
                                   logit_scale, biasbuf, scalebuf);
  attn_kernel<<<4096, 256, 0, stream>>>(X, wbf, bq, bv, bo, biasbuf, scalebuf,
                                        out);
}
